// CausalMolSSM_73366631350462
// MI455X (gfx1250) — matrix operand order repaired and checked by reading, not compiled, not run on hardware
//
#include <hip/hip_runtime.h>
#include <hip/hip_bf16.h>

// ---------------------------------------------------------------------------
// CausalMolSSM on MI455X (gfx1250, wave32).
//   * 4 GEMMs via v_wmma_f32_16x16x32_f16 (one wave -> 16x64 output tile,
//     K-loop software-pipelined / double-buffered)
//   * depthwise conv4 + SiLU fused elementwise
//   * complex selective scan as direct recurrence, 16 states = 16 lanes,
//     2 channels per wave32, shfl_xor tree-reduce inside 16-lane halves
// ---------------------------------------------------------------------------

typedef __attribute__((ext_vector_type(16))) _Float16 v16h;
typedef __attribute__((ext_vector_type(8)))  _Float16 v8h;
typedef __attribute__((ext_vector_type(8)))  float    v8f;

#define D_MODEL 1024
#define D_STATE 16
#define D_INNER 2048
#define SEQ_L   1024
#define N_SSM   (D_INNER + 4 * D_STATE)   // 2112
#define N_XZ    (2 * D_INNER)             // 4096

// ---------------------------------------------------------------- converts
__global__ __launch_bounds__(256) void f32_to_f16_kernel(const float* __restrict__ src,
                                                         _Float16* __restrict__ dst, int n) {
  int i = blockIdx.x * 256 + threadIdx.x;
  if (i < n) dst[i] = (_Float16)src[i];
}

__global__ __launch_bounds__(256) void f32_to_f16_strided_kernel(const float* __restrict__ src,
                                                                 int src_ld,
                                                                 _Float16* __restrict__ dst,
                                                                 int dst_ld, int rows, int cols) {
  int i = blockIdx.x * 256 + threadIdx.x;
  if (i >= rows * cols) return;
  int r = i / cols, c = i - r * cols;
  dst[(size_t)r * dst_ld + c] = (_Float16)src[(size_t)r * src_ld + c];
}

// ---------------------------------------------------------------- WMMA GEMM
// C[M,N] = act( A[M,K] @ W[N,K]^T + bias )   (A,W f16 row-major; C f32)
// grid = (M/16, N/64), block = 32 (one wave). K % 32 == 0, K >= 64.
//
// Fragment addressing used here, same contiguous K order for A and B (16-bit A 16x32, B 32x16,
// 32-bit C 16x16):
//   A: lane l holds row (l&15), 16 contiguous K at (l>>4)*16
//   B: lane l holds weight row (=output col) (l&15), 16 contiguous K at (l>>4)*16
//   C: vgpr r -> row r + 8*(l>>4), col (l&15)
template <int ACT>   // 0 = none, 1 = softplus
__global__ __launch_bounds__(32) void gemm_wmma_f16_kernel(
    const _Float16* __restrict__ A, const _Float16* __restrict__ W,
    const float* __restrict__ bias, float* __restrict__ C,
    int M, int N, int K) {
  const int lane = threadIdx.x;
  const int row  = lane & 15;
  const int hk   = (lane >> 4) * 16;   // A half-K select
  const int bk   = (lane >> 4) * 16;   // B half-K select

  const _Float16* aRow = A + ((size_t)(blockIdx.x * 16 + row)) * K;
  const _Float16* wRow = W + ((size_t)(blockIdx.y * 64 + row)) * K;

  auto loadA = [&](int k) -> v16h {
    return *(const v16h*)(aRow + k + hk);
  };
  auto loadB = [&](int k, int j) -> v16h {
    return *(const v16h*)(wRow + (size_t)(j * 16) * K + k + bk);
  };

  v8f acc0 = {}, acc1 = {}, acc2 = {}, acc3 = {};

  // --- software-pipelined K loop: next fragments load while current WMMAs run
  v16h a  = loadA(0);
  v16h b0 = loadB(0, 0), b1 = loadB(0, 1), b2 = loadB(0, 2), b3 = loadB(0, 3);
  for (int k = 0; k < K - 32; k += 32) {
    const int kn = k + 32;
    __builtin_prefetch(aRow + k + 256, 0, 0);           // global_prefetch_b8
    __builtin_prefetch(wRow + k + 256, 0, 0);
    v16h an  = loadA(kn);
    v16h bn0 = loadB(kn, 0);
    v16h bn1 = loadB(kn, 1);
    v16h bn2 = loadB(kn, 2);
    v16h bn3 = loadB(kn, 3);
    acc0 = __builtin_amdgcn_wmma_f32_16x16x32_f16(false, a, false, b0, (short)0, acc0, false, false);
    acc1 = __builtin_amdgcn_wmma_f32_16x16x32_f16(false, a, false, b1, (short)0, acc1, false, false);
    acc2 = __builtin_amdgcn_wmma_f32_16x16x32_f16(false, a, false, b2, (short)0, acc2, false, false);
    acc3 = __builtin_amdgcn_wmma_f32_16x16x32_f16(false, a, false, b3, (short)0, acc3, false, false);
    a = an; b0 = bn0; b1 = bn1; b2 = bn2; b3 = bn3;
  }
  acc0 = __builtin_amdgcn_wmma_f32_16x16x32_f16(false, a, false, b0, (short)0, acc0, false, false);
  acc1 = __builtin_amdgcn_wmma_f32_16x16x32_f16(false, a, false, b1, (short)0, acc1, false, false);
  acc2 = __builtin_amdgcn_wmma_f32_16x16x32_f16(false, a, false, b2, (short)0, acc2, false, false);
  acc3 = __builtin_amdgcn_wmma_f32_16x16x32_f16(false, a, false, b3, (short)0, acc3, false, false);

  const int rowbase = blockIdx.x * 16 + (lane >> 4) * 8;
  const int colbase = blockIdx.y * 64 + (lane & 15);
  v8f accs[4] = {acc0, acc1, acc2, acc3};
#pragma unroll
  for (int j = 0; j < 4; ++j) {
    int col = colbase + j * 16;
    float bv = (bias != nullptr) ? bias[col] : 0.f;
#pragma unroll
    for (int r = 0; r < 8; ++r) {
      float v = accs[j][r] + bv;
      if (ACT == 1) v = (v > 20.f) ? v : log1pf(__expf(v));  // softplus
      C[(size_t)(rowbase + r) * N + col] = v;
    }
  }
}

// ---------------------------------------------- depthwise conv4 + bias + SiLU
__global__ __launch_bounds__(256) void conv_silu_kernel(
    const float* __restrict__ xz, const float* __restrict__ cw,
    const float* __restrict__ cb, float* __restrict__ u32,
    _Float16* __restrict__ u16) {
  int i = blockIdx.x * 256 + threadIdx.x;        // i = t*D_INNER + d
  if (i >= SEQ_L * D_INNER) return;
  int t = i / D_INNER, d = i - t * D_INNER;
  const float* w = cw + d * 4;
  float acc = cb[d];
#pragma unroll
  for (int j = 0; j < 4; ++j) {
    int tt = t + j - 3;
    float xv = (tt >= 0) ? xz[(size_t)tt * N_XZ + d] : 0.f;
    acc = fmaf(w[j], xv, acc);
  }
  float s = acc / (1.f + __expf(-acc));          // SiLU
  u32[i] = s;
  u16[i] = (_Float16)s;
}

// ---------------------------------------------------------- complex selective scan
// One wave per 2 channels: lane = 16*half + state. H[t] = Abar*H[t-1] + ubar,
// Abar = (2+dtA)/(2-dtA), ubar = 2*dt*u/(2-dtA) * B.  y = Re(C.H) + D*u, then
// gate with silu(z) and emit f16 for the out_proj WMMA GEMM.
__global__ __launch_bounds__(32) void scan_kernel(
    const float* __restrict__ delta, const float* __restrict__ ssm,
    const float* __restrict__ u, const float* __restrict__ Alr,
    const float* __restrict__ Ali, const float* __restrict__ Dv,
    const float* __restrict__ xz, _Float16* __restrict__ yg) {
  const int lane = threadIdx.x;
  const int half = lane >> 4;
  const int n    = lane & 15;
  const int d    = blockIdx.x * 2 + half;

  float lr  = Alr[d * D_STATE + n];
  float li  = Ali[d * D_STATE + n];
  float mag = __expf(lr);
  float Are = -mag * __cosf(li);
  float Aim = -mag * __sinf(li);
  float Dd  = Dv[d];

  float Hre = 0.f, Him = 0.f;
  for (int t = 0; t < SEQ_L; ++t) {
    float dt = delta[(size_t)t * D_INNER + d];
    float ut = u[(size_t)t * D_INNER + d];
    const float* sr = ssm + (size_t)t * N_SSM + D_INNER;
    float Bre = sr[n], Bim = sr[16 + n], Cre = sr[32 + n], Cim = sr[48 + n];

    float dAre = dt * Are, dAim = dt * Aim;
    float dre = 2.f - dAre, dim = -dAim;        // denom = 2 - dt*A
    float nre = 2.f + dAre, nim = dAim;         // numer = 2 + dt*A
    float inv = 1.f / fmaf(dre, dre, dim * dim);
    float abre = (nre * dre + nim * dim) * inv; // Abar = numer/denom
    float abim = (nim * dre - nre * dim) * inv;
    float s = 2.f * dt * ut * inv;              // 2*dt*u / |denom|^2
    float wre = s * (Bre * dre + Bim * dim);    // ubar = s * B * conj(denom)
    float wim = s * (Bim * dre - Bre * dim);
    float hre = abre * Hre - abim * Him + wre;
    float him = abre * Him + abim * Hre + wim;
    Hre = hre; Him = him;

    float yn = Cre * hre - Cim * him;           // Re(C * H)
    yn += __shfl_xor(yn, 1);                    // reduce within 16-lane half
    yn += __shfl_xor(yn, 2);
    yn += __shfl_xor(yn, 4);
    yn += __shfl_xor(yn, 8);
    if (n == 0) {
      float yv = yn + Dd * ut;
      float zv = xz[(size_t)t * N_XZ + D_INNER + d];
      float sz = zv / (1.f + __expf(-zv));      // silu(z)
      yg[(size_t)t * D_INNER + d] = (_Float16)(yv * sz);
    }
  }
}

// ---------------------------------------------------------------------------
extern "C" void kernel_launch(void* const* d_in, const int* in_sizes, int n_in,
                              void* d_out, int out_size, void* d_ws, size_t ws_size,
                              hipStream_t stream) {
  (void)in_sizes; (void)n_in; (void)out_size; (void)ws_size;
  const float* x          = (const float*)d_in[0];
  const float* in_proj_w  = (const float*)d_in[1];
  const float* conv_w     = (const float*)d_in[2];
  const float* conv_b     = (const float*)d_in[3];
  const float* x_proj_w   = (const float*)d_in[4];
  const float* dt_proj_w  = (const float*)d_in[5];
  const float* dt_proj_b  = (const float*)d_in[6];
  const float* A_log_re   = (const float*)d_in[7];
  const float* A_log_im   = (const float*)d_in[8];
  const float* Dvec       = (const float*)d_in[9];
  const float* out_proj_w = (const float*)d_in[10];
  float* out = (float*)d_out;

  char* ws = (char*)d_ws;
  size_t off = 0;
  auto carve = [&](size_t bytes) -> void* {
    void* p = ws + off;
    off = (off + bytes + 255) & ~(size_t)255;
    return p;
  };
  _Float16* hW  = (_Float16*)carve((size_t)N_SSM * D_INNER * 2);   // f16 weight arena (max 2112x2048)
  _Float16* hA  = (_Float16*)carve((size_t)SEQ_L * D_INNER * 2);   // f16 act arena (x -> u -> delta -> y*silu(z))
  float*    xz  = (float*)carve((size_t)SEQ_L * N_XZ * 4);
  float*    u32 = (float*)carve((size_t)SEQ_L * D_INNER * 4);
  float*    ssm = (float*)carve((size_t)SEQ_L * N_SSM * 4);
  float*    dlt = (float*)carve((size_t)SEQ_L * D_INNER * 4);

  auto cvt = [&](const float* s, _Float16* dst, int n) {
    f32_to_f16_kernel<<<(n + 255) / 256, 256, 0, stream>>>(s, dst, n);
  };

  // 1) xz = x @ in_proj_w.T                  M=1024 K=1024 N=4096
  cvt(x, hA, SEQ_L * D_MODEL);
  cvt(in_proj_w, hW, N_XZ * D_MODEL);
  gemm_wmma_f16_kernel<0><<<dim3(SEQ_L / 16, N_XZ / 64), 32, 0, stream>>>(
      hA, hW, nullptr, xz, SEQ_L, N_XZ, D_MODEL);

  // 2) u = silu(depthwise_conv4(xc) + conv_b)
  conv_silu_kernel<<<(SEQ_L * D_INNER + 255) / 256, 256, 0, stream>>>(
      xz, conv_w, conv_b, u32, hA);

  // 3) ssm = u @ x_proj_w.T                  M=1024 K=2048 N=2112
  cvt(x_proj_w, hW, N_SSM * D_INNER);
  gemm_wmma_f16_kernel<0><<<dim3(SEQ_L / 16, N_SSM / 64), 32, 0, stream>>>(
      hA, hW, nullptr, ssm, SEQ_L, N_SSM, D_INNER);

  // 4) delta = softplus(ssm[:, :2048] @ dt_proj_w.T + dt_proj_b)
  f32_to_f16_strided_kernel<<<(SEQ_L * D_INNER + 255) / 256, 256, 0, stream>>>(
      ssm, N_SSM, hA, D_INNER, SEQ_L, D_INNER);
  cvt(dt_proj_w, hW, D_INNER * D_INNER);
  gemm_wmma_f16_kernel<1><<<dim3(SEQ_L / 16, D_INNER / 64), 32, 0, stream>>>(
      hA, hW, dt_proj_b, dlt, SEQ_L, D_INNER, D_INNER);

  // 5) selective scan + D*u + silu(z) gate -> yg (f16)
  scan_kernel<<<D_INNER / 2, 32, 0, stream>>>(
      dlt, ssm, u32, A_log_re, A_log_im, Dvec, xz, hA);

  // 6) out = yg @ out_proj_w.T               M=1024 K=2048 N=1024
  cvt(out_proj_w, hW, D_MODEL * D_INNER);
  gemm_wmma_f16_kernel<0><<<dim3(SEQ_L / 16, D_MODEL / 64), 32, 0, stream>>>(
      hA, hW, nullptr, out, SEQ_L, D_MODEL, D_INNER);
}
